// DynamicHybridAttention_50714973831340
// MI455X (gfx1250) — matrix operand order repaired and checked by reading, not compiled, not run on hardware
//
#include <hip/hip_runtime.h>
#include <hip/hip_bf16.h>

// ---------------------------------------------------------------------------
// DynamicHybridAttention for MI455X (gfx1250, wave32).
// All GEMMs via v_wmma_f32_16x16x32_bf16; guard-free GEMMs use async
// global->LDS staging (ASYNCcnt) with ping-pong LDS buffers; edge-tiled
// GEMMs (416-sized attention) use predicated sync staging with zero fill.
// ---------------------------------------------------------------------------

#define B_ 8
#define N_ 4096
#define C_ 512
#define H_ 8
#define DH_ 64
#define TED_ 1280
#define INTER_ 512
#define KSEL_ 409            // max(16, int(4096*0.1))
#define KP_ 416              // KSEL padded to multiple of 16
#define BT_ (B_ * KP_)       // 3328 routed-token rows (padded)
#define BH_ (B_ * H_)        // 64 attention batches
#define TILE_ (128 * 32)     // one LDS tile in bf16 elements

typedef __bf16 bf16_t;
typedef bf16_t v16bf __attribute__((ext_vector_type(16)));
typedef float  v8f   __attribute__((ext_vector_type(8)));
typedef unsigned int v4u __attribute__((ext_vector_type(4)));

#if defined(__has_builtin)
#if __has_builtin(__builtin_amdgcn_global_load_async_to_lds_b128)
#define DHA_ASYNC_BUILTIN 1
#endif
#if __has_builtin(__builtin_amdgcn_s_wait_asynccnt)
#define DHA_WAIT_BUILTIN 1
#endif
#endif

// ---------------- helpers ----------------

__device__ inline unsigned short f2bf(float f) {
  unsigned u = __float_as_uint(f);
  unsigned r = u + 0x7FFFu + ((u >> 16) & 1u);  // round-to-nearest-even
  return (unsigned short)(r >> 16);
}
__device__ inline float bf2f(unsigned short h) {
  return __uint_as_float(((unsigned)h) << 16);
}

// one 16B lane-chunk, global -> LDS via the CDNA5 async data path (ASYNCcnt)
__device__ inline void async_ld128(const unsigned short* g, unsigned short* l) {
#if defined(DHA_ASYNC_BUILTIN)
  typedef int v4i_vs __attribute__((vector_size(16)));
  typedef __attribute__((address_space(1))) v4i_vs* gp_t;   // global int4*
  typedef __attribute__((address_space(3))) v4i_vs* lp_t;   // LDS int4*
  __builtin_amdgcn_global_load_async_to_lds_b128((gp_t)g, (lp_t)l, 0, 0);
#else
  // LDS byte address = low 32 bits of the generic address (ISA 10.2)
  unsigned la = (unsigned)(size_t)l;
  asm volatile("global_load_async_to_lds_b128 %0, %1, off"
               :: "v"(la), "v"(g) : "memory");
#endif
}
__device__ inline void wait_async() {
#if defined(DHA_WAIT_BUILTIN)
  __builtin_amdgcn_s_wait_asynccnt(0);
#else
  asm volatile("s_wait_asynccnt 0x0" ::: "memory");
#endif
}

union FragAB { v4u u[2]; v16bf v; };

// A fragment (16x32 bf16): lane<16 -> row M=lane, K {0..7,16..23};
// lane>=16 -> row M=lane-16, K {8..15,24..31}.  LDS tile is [row][32k].
__device__ inline v16bf ldsA_frag(const unsigned short* s, int mbase, int lane) {
  int lr = lane & 15, hi = lane >> 4;
  const unsigned short* p = s + (mbase + lr) * 32 + hi * 8;
  FragAB f;
  f.u[0] = *(const v4u*)p;
  f.u[1] = *(const v4u*)(p + 16);
  return f.v;
}
// B fragment (32x16 bf16, LDS tile stored [n][32k]): lane<16 -> N=lane,
// K {0..7,16..23}; lane>=16 -> N=lane-16, K {8..15,24..31}.
__device__ inline v16bf ldsB_frag(const unsigned short* s, int nbase, int lane) {
  int lr = lane & 15, hi = lane >> 4;
  const unsigned short* p = s + (nbase + lr) * 32 + hi * 8;
  FragAB f;
  f.u[0] = *(const v4u*)p;
  f.u[1] = *(const v4u*)(p + 16);
  return f.v;
}

__device__ inline v8f wmma_bf16(v16bf a, v16bf b, v8f c) {
  return __builtin_amdgcn_wmma_f32_16x16x32_bf16(false, a, false, b,
                                                 (short)0, c, false, false);
}

// ---------------- shared 128x128 GEMM core (8 waves, wave = 64Mx32N) -------
// A: [M][K] bf16 row-major (lda), B: [N][K] bf16 row-major (ldb).

__device__ inline void compute_tile(const unsigned short* sA, const unsigned short* sB,
                                    v8f acc[4][2], int lane, int wm, int wn) {
  v16bf b0 = ldsB_frag(sB, wn, lane);
  v16bf b1 = ldsB_frag(sB, wn + 16, lane);
#pragma unroll
  for (int mf = 0; mf < 4; ++mf) {
    v16bf a = ldsA_frag(sA, wm + mf * 16, lane);
    acc[mf][0] = wmma_bf16(a, b0, acc[mf][0]);
    acc[mf][1] = wmma_bf16(a, b1, acc[mf][1]);
  }
}

// guard-free async staging: 2 x b128 for A and B per thread (8KB each tile)
__device__ inline void stage_tile_async(const unsigned short* A, const unsigned short* Bm,
                                        int lda, int ldb, int mBase, int nBase, int kt,
                                        unsigned short* sA, unsigned short* sB, int tid) {
#pragma unroll
  for (int r = 0; r < 2; ++r) {
    int cidx = tid + r * 256;
    int row = cidx >> 2, seg = cidx & 3;
    int gk = kt + seg * 8;
    async_ld128(A + (long)(mBase + row) * lda + gk, sA + row * 32 + seg * 8);
    async_ld128(Bm + (long)(nBase + row) * ldb + gk, sB + row * 32 + seg * 8);
  }
}

__device__ inline void gemm_core(const unsigned short* A, const unsigned short* Bm,
                                 int M, int N, int K, int lda, int ldb,
                                 int mBlk, int nBlk,
                                 unsigned short* sA2, unsigned short* sB2,
                                 v8f acc[4][2], int tid) {
  int lane = tid & 31, wid = tid >> 5;
  int wm = (wid >> 2) * 64, wn = (wid & 3) * 32;
  bool full = ((M & 127) == 0) && ((N & 127) == 0) && ((K & 31) == 0);
  if (full) {
    // async double-buffered pipeline: fill tile kt+32 while computing kt
    int buf = 0;
    stage_tile_async(A, Bm, lda, ldb, mBlk * 128, nBlk * 128, 0, sA2, sB2, tid);
    wait_async();
    __syncthreads();
    for (int kt = 0; kt < K; kt += 32) {
      if (kt + 32 < K)
        stage_tile_async(A, Bm, lda, ldb, mBlk * 128, nBlk * 128, kt + 32,
                         sA2 + (buf ^ 1) * TILE_, sB2 + (buf ^ 1) * TILE_, tid);
      compute_tile(sA2 + buf * TILE_, sB2 + buf * TILE_, acc, lane, wm, wn);
      if (kt + 32 < K) {
        wait_async();
        __syncthreads();
      }
      buf ^= 1;
    }
  } else {
    // predicated sync path with zero fill (edge tiles: 416-sized dims)
    for (int kt = 0; kt < K; kt += 32) {
      __syncthreads();
#pragma unroll
      for (int r = 0; r < 2; ++r) {
        int cidx = tid + r * 256;
        int row = cidx >> 2, seg = cidx & 3;
        int gk = kt + seg * 8;
        v4u za = {0u, 0u, 0u, 0u};
        v4u zb = {0u, 0u, 0u, 0u};
        int gm = mBlk * 128 + row;
        if (gm < M && gk < K) za = *(const v4u*)(A + (long)gm * lda + gk);
        *(v4u*)(sA2 + row * 32 + seg * 8) = za;
        int gn = nBlk * 128 + row;
        if (gn < N && gk < K) zb = *(const v4u*)(Bm + (long)gn * ldb + gk);
        *(v4u*)(sB2 + row * 32 + seg * 8) = zb;
      }
      __syncthreads();
      compute_tile(sA2, sB2, acc, lane, wm, wn);
    }
  }
}

// C/D layout: v8f element r of lane: (M = wm+mf*16 + (lane>>4)*8 + r,
//                                     N = wn+nf*16 + (lane&15))
__device__ inline void gemm_epilogue(v8f acc[4][2], int M, int N, int ldc,
                                     int mBlk, int nBlk, void* out, float scale,
                                     const float* bias, int out_bf16, int tid) {
  int lane = tid & 31, wid = tid >> 5;
  int wm = (wid >> 2) * 64, wn = (wid & 3) * 32;
  int lr = lane & 15, hi = lane >> 4;
#pragma unroll
  for (int mf = 0; mf < 4; ++mf) {
#pragma unroll
    for (int nf = 0; nf < 2; ++nf) {
      int n = nBlk * 128 + wn + nf * 16 + lr;
      if (n >= N) continue;
      float bv = bias ? bias[n] : 0.0f;
#pragma unroll
      for (int r = 0; r < 8; ++r) {
        int m = mBlk * 128 + wm + mf * 16 + hi * 8 + r;
        if (m >= M) continue;
        float val = acc[mf][nf][r] * scale + bv;
        if (out_bf16) ((unsigned short*)out)[(long)m * ldc + n] = f2bf(val);
        else          ((float*)out)[(long)m * ldc + n] = val;
      }
    }
  }
}

// ---------------- kernels ----------------

__global__ __launch_bounds__(256) void dha_gemm_bf16(
    const unsigned short* A, const unsigned short* Bm, void* out,
    int M, int N, int K, int lda, int ldb, int ldc,
    float scale, const float* bias, int out_bf16) {
  __shared__ unsigned short sA[2 * TILE_];
  __shared__ unsigned short sB[2 * TILE_];
  v8f z = {0, 0, 0, 0, 0, 0, 0, 0};
  v8f acc[4][2] = {{z, z}, {z, z}, {z, z}, {z, z}};
  int tid = threadIdx.x;
  gemm_core(A, Bm, M, N, K, lda, ldb, blockIdx.y, blockIdx.x, sA, sB, acc, tid);
  gemm_epilogue(acc, M, N, ldc, blockIdx.y, blockIdx.x, out, scale, bias,
                out_bf16, tid);
}

// batched sim = q @ k^T * DH^-0.5 per (b,h)
__global__ __launch_bounds__(256) void dha_gemm_sim(
    const unsigned short* qb, const unsigned short* kb, float* sim) {
  __shared__ unsigned short sA[2 * TILE_];
  __shared__ unsigned short sB[2 * TILE_];
  int z = blockIdx.z, b = z >> 3, h = z & 7;
  const unsigned short* A = qb + ((long)b * KP_) * C_ + h * DH_;
  const unsigned short* Bm = kb + ((long)b * KP_) * C_ + h * DH_;
  float* out = sim + (long)z * KP_ * KP_;
  v8f zv = {0, 0, 0, 0, 0, 0, 0, 0};
  v8f acc[4][2] = {{zv, zv}, {zv, zv}, {zv, zv}, {zv, zv}};
  int tid = threadIdx.x;
  gemm_core(A, Bm, KP_, KP_, DH_, C_, C_, blockIdx.y, blockIdx.x, sA, sB, acc, tid);
  gemm_epilogue(acc, KP_, KP_, KP_, blockIdx.y, blockIdx.x, out, 0.125f,
                nullptr, 0, tid);
}

// batched sa_h = p @ v  per (b,h), v pre-transposed to [d][j]
__global__ __launch_bounds__(256) void dha_gemm_pv(
    const unsigned short* p, const unsigned short* vT, unsigned short* sab) {
  __shared__ unsigned short sA[2 * TILE_];
  __shared__ unsigned short sB[2 * TILE_];
  int z = blockIdx.z, b = z >> 3, h = z & 7;
  const unsigned short* A = p + (long)z * KP_ * KP_;
  const unsigned short* Bm = vT + (long)z * DH_ * KP_;
  unsigned short* out = sab + ((long)b * KP_) * C_ + h * DH_;
  v8f zv = {0, 0, 0, 0, 0, 0, 0, 0};
  v8f acc[4][2] = {{zv, zv}, {zv, zv}, {zv, zv}, {zv, zv}};
  int tid = threadIdx.x;
  gemm_core(A, Bm, KP_, DH_, KP_, KP_, KP_, blockIdx.y, blockIdx.x, sA, sB, acc, tid);
  gemm_epilogue(acc, KP_, DH_, C_, blockIdx.y, blockIdx.x, out, 1.0f,
                nullptr, 1, tid);
}

// sa @ wo: epilogue scatter-adds alpha*(acc+bo) into out at top-k rows
__global__ __launch_bounds__(256) void dha_gemm_scatter(
    const unsigned short* sab, const unsigned short* woT, float* out,
    const float* bo, const float* alphaPtr, const int* idx) {
  __shared__ unsigned short sA[2 * TILE_];
  __shared__ unsigned short sB[2 * TILE_];
  v8f zv = {0, 0, 0, 0, 0, 0, 0, 0};
  v8f acc[4][2] = {{zv, zv}, {zv, zv}, {zv, zv}, {zv, zv}};
  int tid = threadIdx.x;
  gemm_core(sab, woT, BT_, C_, C_, C_, C_, blockIdx.y, blockIdx.x, sA, sB, acc, tid);
  float alpha = alphaPtr[0];
  int lane = tid & 31, wid = tid >> 5;
  int wm = (wid >> 2) * 64, wn = (wid & 3) * 32;
  int lr = lane & 15, hi = lane >> 4;
#pragma unroll
  for (int mf = 0; mf < 4; ++mf) {
#pragma unroll
    for (int nf = 0; nf < 2; ++nf) {
      int n = blockIdx.x * 128 + wn + nf * 16 + lr;
      float bv = bo[n];
#pragma unroll
      for (int r = 0; r < 8; ++r) {
        int m = blockIdx.y * 128 + wm + mf * 16 + hi * 8 + r;
        int b = m / KP_, t = m - b * KP_;
        if (t < KSEL_) {
          int row = idx[b * KP_ + t];
          float* po = out + ((long)b * N_ + row) * C_ + n;
          *po += alpha * (acc[mf][nf][r] + bv);   // unique writer per element
        }
      }
    }
  }
}

// gating MLP: scores[b,c]
__global__ __launch_bounds__(256) void dha_gating(
    const float* temb, const float* gw1, const float* gb1,
    const float* gw2, const float* gb2, float* scores) {
  __shared__ float hid[64];
  int b = blockIdx.x, tid = threadIdx.x;
  if (tid < 64) {
    float s = gb1[tid];
    const float* tr = temb + (long)b * TED_;
    for (int t = 0; t < TED_; ++t) s += tr[t] * gw1[t * 64 + tid];
    hid[tid] = s / (1.0f + __expf(-s));  // silu
  }
  __syncthreads();
  for (int c = tid; c < C_; c += 256) {
    float s = gb2[c];
    for (int j = 0; j < 64; ++j) s += hid[j] * gw2[j * C_ + c];
    scores[b * C_ + c] = s;
  }
}

// imp[b,n] = dot(x[b,n,:], scores[b,:]) -- one wave per row
__global__ __launch_bounds__(256) void dha_imp(
    const float* x, const float* scores, float* imp) {
  int tid = threadIdx.x, lane = tid & 31, wid = tid >> 5;
  int row = blockIdx.x * 8 + wid;
  int b = row >> 12;
  const float* xr = x + (long)row * C_;
  const float* sc = scores + b * C_;
  float s = 0.f;
  for (int c = lane; c < C_; c += 32) s += xr[c] * sc[c];
  for (int off = 16; off > 0; off >>= 1) s += __shfl_xor(s, off, 32);
  if (lane == 0) imp[row] = s;
}

// per-batch bitonic top-k: sort 4096 packed keys ascending, take first 409
__global__ __launch_bounds__(1024) void dha_topk(const float* imp, int* idx) {
  __shared__ unsigned long long key[N_];
  int b = blockIdx.x, tid = threadIdx.x;
  for (int n = tid; n < N_; n += 1024) {
    unsigned u = __float_as_uint(imp[b * N_ + n]);
    unsigned ord = (u >> 31) ? ~u : (u | 0x80000000u);
    key[n] = (((unsigned long long)(ord ^ 0xFFFFFFFFu)) << 32) | (unsigned)n;
  }
  __syncthreads();
  for (int size = 2; size <= N_; size <<= 1) {
    for (int stride = size >> 1; stride > 0; stride >>= 1) {
      for (int t = tid; t < N_ / 2; t += 1024) {
        int pos = 2 * t - (t & (stride - 1));
        int j = pos + stride;
        bool asc = ((pos & size) == 0);
        unsigned long long a = key[pos], c = key[j];
        if ((a > c) == asc) { key[pos] = c; key[j] = a; }
      }
      __syncthreads();
    }
  }
  for (int t = tid; t < KSEL_; t += 1024)
    idx[b * KP_ + t] = (int)(key[t] & 0xFFFFFFFFu);
}

// gather routed tokens -> bf16 (pad rows zeroed)
__global__ __launch_bounds__(256) void dha_gather(
    const float* x, const int* idx, unsigned short* ximp) {
  long g = (long)blockIdx.x * 256 + threadIdx.x;
  if (g >= (long)BT_ * C_) return;
  int b = (int)(g / ((long)KP_ * C_));
  int rem = (int)(g - (long)b * KP_ * C_);
  int t = rem >> 9, c = rem & (C_ - 1);
  float val = 0.f;
  if (t < KSEL_) {
    int row = idx[b * KP_ + t];
    val = x[((long)b * N_ + row) * C_ + c];
  }
  ximp[g] = f2bf(val);
}

// 512x512 transpose + cast (weights -> [n][k] bf16)
__global__ __launch_bounds__(256) void dha_castT(const float* src, unsigned short* dst) {
  int g = blockIdx.x * 256 + threadIdx.x;
  int k = g >> 9, n = g & 511;
  dst[n * 512 + k] = f2bf(src[g]);
}
__global__ __launch_bounds__(256) void dha_cast(const float* src, unsigned short* dst) {
  int g = blockIdx.x * 256 + threadIdx.x;
  dst[g] = f2bf(src[g]);
}

// LayerNorm over C -> bf16
__global__ __launch_bounds__(256) void dha_ln(
    const float* x, const float* lng, const float* lnb, unsigned short* xn) {
  __shared__ float r1[256], r2[256];
  long row = blockIdx.x;
  int tid = threadIdx.x;
  const float* xr = x + row * C_;
  float v0 = xr[tid], v1 = xr[tid + 256];
  r1[tid] = v0 + v1;
  r2[tid] = v0 * v0 + v1 * v1;
  __syncthreads();
  for (int off = 128; off > 0; off >>= 1) {
    if (tid < off) { r1[tid] += r1[tid + off]; r2[tid] += r2[tid + off]; }
    __syncthreads();
  }
  float mean = r1[0] * (1.0f / C_);
  float var = r2[0] * (1.0f / C_) - mean * mean;
  float rstd = rsqrtf(var + 1e-5f);
  xn[row * C_ + tid]       = f2bf((v0 - mean) * rstd * lng[tid] + lnb[tid]);
  xn[row * C_ + tid + 256] = f2bf((v1 - mean) * rstd * lng[tid + 256] + lnb[tid + 256]);
}

// online column softmax stats over N per (b, i): colmax/colsum of s=e*0.125
__global__ __launch_bounds__(256) void dha_colstats(
    const unsigned short* e, float* colmax, float* colsum) {
  __shared__ float sm[256], ss[256];
  int blk = blockIdx.x, b = blk >> 3, i0 = (blk & 7) * 64;
  int tid = threadIdx.x, il = tid & 63, ns = tid >> 6;
  float m = -3.0e38f, s = 0.f;
  for (int n = ns; n < N_; n += 4) {
    float v = bf2f(e[((long)b * N_ + n) * INTER_ + i0 + il]);
    if (v > m) { s = s * __expf(m - v) + 1.0f; m = v; }
    else s += __expf(v - m);
  }
  sm[tid] = m; ss[tid] = s;
  __syncthreads();
  for (int off = 128; off >= 64; off >>= 1) {
    if (tid < off) {
      float m1 = sm[tid], s1 = ss[tid], m2 = sm[tid + off], s2 = ss[tid + off];
      float M = fmaxf(m1, m2);
      sm[tid] = M;
      ss[tid] = s1 * __expf(m1 - M) + s2 * __expf(m2 - M);
    }
    __syncthreads();
  }
  if (tid < 64) {
    colmax[b * INTER_ + i0 + tid] = sm[tid];
    colsum[b * INTER_ + i0 + tid] = ss[tid];
  }
}

// a[b,n,i] = (p/colsum) / (per-head sum + 1e-6), bf16
__global__ __launch_bounds__(256) void dha_a(
    const unsigned short* e, const float* colmax, const float* colsum,
    unsigned short* a) {
  __shared__ float p[INTER_];
  __shared__ float den[H_];
  long row = blockIdx.x;
  int b = blockIdx.x >> 12;
  int tid = threadIdx.x;
  const unsigned short* er = e + row * INTER_;
  for (int i = tid; i < INTER_; i += 256) {
    float v = bf2f(er[i]);
    p[i] = __expf(v - colmax[b * INTER_ + i]) / colsum[b * INTER_ + i];
  }
  __syncthreads();
  if (tid < H_) {
    float s = 0.f;
    for (int j = 0; j < 64; ++j) s += p[tid * 64 + j];
    den[tid] = s + 1e-6f;
  }
  __syncthreads();
  for (int i = tid; i < INTER_; i += 256)
    a[row * INTER_ + i] = f2bf(p[i] / den[i >> 6]);
}

// row softmax over j (j >= KSEL masked out), bf16 p (pads -> 0)
__global__ __launch_bounds__(256) void dha_softmax(const float* sim, unsigned short* p) {
  __shared__ float red[256];
  long row = blockIdx.x;
  int tid = threadIdx.x;
  const float* sr = sim + row * KP_;
  float v0 = (tid < KSEL_) ? sr[tid] : -3.0e38f;
  float v1 = (tid + 256 < KSEL_) ? sr[tid + 256] : -3.0e38f;
  red[tid] = fmaxf(v0, v1);
  __syncthreads();
  for (int off = 128; off > 0; off >>= 1) {
    if (tid < off) red[tid] = fmaxf(red[tid], red[tid + off]);
    __syncthreads();
  }
  float m = red[0];
  __syncthreads();
  float e0 = (tid < KSEL_) ? __expf(v0 - m) : 0.f;
  float e1 = (tid + 256 < KSEL_) ? __expf(v1 - m) : 0.f;
  red[tid] = e0 + e1;
  __syncthreads();
  for (int off = 128; off > 0; off >>= 1) {
    if (tid < off) red[tid] += red[tid + off];
    __syncthreads();
  }
  float inv = 1.0f / red[0];
  p[row * KP_ + tid] = f2bf(e0 * inv);
  if (tid + 256 < KP_) p[row * KP_ + tid + 256] = f2bf(e1 * inv);
}

// vT[z][d][j] = v[b][j][h*64+d]  (bf16 copy with transpose, z = b*H+h)
__global__ __launch_bounds__(256) void dha_vt(const unsigned short* vb, unsigned short* vT) {
  long g = (long)blockIdx.x * 256 + threadIdx.x;
  if (g >= (long)BH_ * DH_ * KP_) return;
  int j = (int)(g % KP_);
  long rest = g / KP_;
  int d = (int)(rest % DH_);
  int z = (int)(rest / DH_);
  int b = z >> 3, h = z & 7;
  vT[g] = vb[((long)b * KP_ + j) * C_ + h * DH_ + d];
}

// ---------------- host ----------------

extern "C" void kernel_launch(void* const* d_in, const int* in_sizes, int n_in,
                              void* d_out, int out_size, void* d_ws, size_t ws_size,
                              hipStream_t stream) {
  (void)in_sizes; (void)n_in; (void)out_size; (void)ws_size;
  const float* x    = (const float*)d_in[0];
  const float* temb = (const float*)d_in[1];
  const float* gw1  = (const float*)d_in[2];
  const float* gb1  = (const float*)d_in[3];
  const float* gw2  = (const float*)d_in[4];
  const float* gb2  = (const float*)d_in[5];
  const float* wq   = (const float*)d_in[6];
  const float* wk   = (const float*)d_in[7];
  const float* wv   = (const float*)d_in[8];
  const float* wo   = (const float*)d_in[9];
  const float* bo   = (const float*)d_in[10];
  const float* lng  = (const float*)d_in[11];
  const float* lnb  = (const float*)d_in[12];
  const float* eak  = (const float*)d_in[13];
  const float* eav  = (const float*)d_in[14];
  const float* alp  = (const float*)d_in[15];
  float* out = (float*)d_out;

  char* ws = (char*)d_ws;
  size_t off = 0;
  auto alloc = [&](size_t bytes) {
    size_t o = off;
    off = (off + bytes + 255) & ~(size_t)255;
    return o;
  };
  size_t o_scores = alloc((size_t)B_ * C_ * 4);
  size_t o_imp    = alloc((size_t)B_ * N_ * 4);
  size_t o_idx    = alloc((size_t)B_ * KP_ * 4);
  size_t o_ximp   = alloc((size_t)BT_ * C_ * 2);
  size_t o_wqT  = alloc((size_t)C_ * C_ * 2);
  size_t o_wkT  = alloc((size_t)C_ * C_ * 2);
  size_t o_wvT  = alloc((size_t)C_ * C_ * 2);
  size_t o_woT  = alloc((size_t)C_ * C_ * 2);
  size_t o_eakb = alloc((size_t)INTER_ * C_ * 2);
  size_t o_eavb = alloc((size_t)C_ * INTER_ * 2);
  size_t o_qb = alloc((size_t)BT_ * C_ * 2);
  size_t o_kb = alloc((size_t)BT_ * C_ * 2);
  size_t o_vb = alloc((size_t)BT_ * C_ * 2);
  size_t o_vT  = alloc((size_t)BH_ * DH_ * KP_ * 2);
  size_t o_sab = alloc((size_t)BT_ * C_ * 2);
  size_t o_cmax = alloc((size_t)B_ * INTER_ * 4);
  size_t o_csum = alloc((size_t)B_ * INTER_ * 4);
  // big aliased union: EA phase {xn, e}, then {a over xn}, then SA {sim, p}
  size_t xnBytes  = (size_t)B_ * N_ * INTER_ * 2;       // 33.5 MB
  size_t simBytes = (size_t)BH_ * KP_ * KP_ * 4;        // 44.3 MB
  size_t pBytes   = (size_t)BH_ * KP_ * KP_ * 2;        // 22.2 MB
  size_t uBytes = 2 * xnBytes;
  if (simBytes + pBytes > uBytes) uBytes = simBytes + pBytes;
  size_t o_U = alloc(uBytes);

  float* scores = (float*)(ws + o_scores);
  float* imp    = (float*)(ws + o_imp);
  int*   idx    = (int*)(ws + o_idx);
  unsigned short* ximp = (unsigned short*)(ws + o_ximp);
  unsigned short* wqT  = (unsigned short*)(ws + o_wqT);
  unsigned short* wkT  = (unsigned short*)(ws + o_wkT);
  unsigned short* wvT  = (unsigned short*)(ws + o_wvT);
  unsigned short* woT  = (unsigned short*)(ws + o_woT);
  unsigned short* eakb = (unsigned short*)(ws + o_eakb);
  unsigned short* eavb = (unsigned short*)(ws + o_eavb);
  unsigned short* qb = (unsigned short*)(ws + o_qb);
  unsigned short* kb = (unsigned short*)(ws + o_kb);
  unsigned short* vb = (unsigned short*)(ws + o_vb);
  unsigned short* vT  = (unsigned short*)(ws + o_vT);
  unsigned short* sab = (unsigned short*)(ws + o_sab);
  float* cmax = (float*)(ws + o_cmax);
  float* csum = (float*)(ws + o_csum);
  unsigned short* xn = (unsigned short*)(ws + o_U);
  unsigned short* eS = (unsigned short*)(ws + o_U + xnBytes);
  unsigned short* aB = (unsigned short*)(ws + o_U);             // aliases xn (dead)
  float* sim = (float*)(ws + o_U);                              // aliases xn/a (dead)
  unsigned short* pB = (unsigned short*)(ws + o_U + simBytes);  // aliases e (dead)

  // --- routing ---
  dha_gating<<<B_, 256, 0, stream>>>(temb, gw1, gb1, gw2, gb2, scores);
  dha_imp<<<(B_ * N_) / 8, 256, 0, stream>>>(x, scores, imp);
  dha_topk<<<B_, 1024, 0, stream>>>(imp, idx);
  dha_gather<<<(int)(((long)BT_ * C_ + 255) / 256), 256, 0, stream>>>(x, idx, ximp);

  // --- weight casts ---
  dha_castT<<<(C_ * C_) / 256, 256, 0, stream>>>(wq, wqT);
  dha_castT<<<(C_ * C_) / 256, 256, 0, stream>>>(wk, wkT);
  dha_castT<<<(C_ * C_) / 256, 256, 0, stream>>>(wv, wvT);
  dha_castT<<<(C_ * C_) / 256, 256, 0, stream>>>(wo, woT);
  dha_cast<<<(INTER_ * C_) / 256, 256, 0, stream>>>(eak, eakb);  // already [i][c]=[n][k]
  dha_cast<<<(C_ * INTER_) / 256, 256, 0, stream>>>(eav, eavb);  // already [c][i]=[n][k]

  // --- external-attention branch (writes full output) ---
  dha_ln<<<B_ * N_, 256, 0, stream>>>(x, lng, lnb, xn);
  dha_gemm_bf16<<<dim3(INTER_ / 128, (B_ * N_) / 128), 256, 0, stream>>>(
      xn, eakb, eS, B_ * N_, INTER_, C_, C_, C_, INTER_, 0.125f, nullptr, 1);
  dha_colstats<<<B_ * (INTER_ / 64), 256, 0, stream>>>(eS, cmax, csum);
  dha_a<<<B_ * N_, 256, 0, stream>>>(eS, cmax, csum, aB);
  dha_gemm_bf16<<<dim3(C_ / 128, (B_ * N_) / 128), 256, 0, stream>>>(
      aB, eavb, out, B_ * N_, C_, INTER_, INTER_, INTER_, C_, 1.0f, nullptr, 0);

  // --- sparse self-attention branch ---
  dha_gemm_bf16<<<dim3(C_ / 128, BT_ / 128), 256, 0, stream>>>(
      ximp, wqT, qb, BT_, C_, C_, C_, C_, C_, 1.0f, nullptr, 1);
  dha_gemm_bf16<<<dim3(C_ / 128, BT_ / 128), 256, 0, stream>>>(
      ximp, wkT, kb, BT_, C_, C_, C_, C_, C_, 1.0f, nullptr, 1);
  dha_gemm_bf16<<<dim3(C_ / 128, BT_ / 128), 256, 0, stream>>>(
      ximp, wvT, vb, BT_, C_, C_, C_, C_, C_, 1.0f, nullptr, 1);
  dha_vt<<<(int)(((long)BH_ * DH_ * KP_ + 255) / 256), 256, 0, stream>>>(vb, vT);
  dha_gemm_sim<<<dim3((KP_ + 127) / 128, (KP_ + 127) / 128, BH_), 256, 0, stream>>>(
      qb, kb, sim);
  dha_softmax<<<BH_ * KP_, 256, 0, stream>>>(sim, pB);
  dha_gemm_pv<<<dim3(1, (KP_ + 127) / 128, BH_), 256, 0, stream>>>(pB, vT, sab);
  dha_gemm_scatter<<<dim3(C_ / 128, BT_ / 128), 256, 0, stream>>>(
      sab, woT, out, bo, alp, idx);
}
